// SentimentRNN_28905129902336
// MI455X (gfx1250) — matrix operand order repaired and checked by reading, not compiled, not run on hardware
//
#include <hip/hip_runtime.h>
#include <hip/hip_bf16.h>
#include <math.h>

typedef __attribute__((ext_vector_type(16))) __bf16 v16bf;
typedef __attribute__((ext_vector_type(8)))  __bf16 v8bf;
typedef __attribute__((ext_vector_type(8)))  float  v8f;
typedef __attribute__((ext_vector_type(4)))  int    v4i;

#define AS1 __attribute__((address_space(1)))
#define AS3 __attribute__((address_space(3)))

#if __has_builtin(__builtin_amdgcn_global_load_async_to_lds_b128) && \
    __has_builtin(__builtin_amdgcn_s_wait_asynccnt)
#define USE_ASYNC_LDS 1
#else
#define USE_ASYNC_LDS 0
#endif

constexpr int Tt  = 2048;   // timesteps
constexpr int Bz  = 128;    // batch
constexpr int Dd  = 100;    // input dim
constexpr int Hh  = 1024;   // hidden dim
constexpr int NBLK = 32;    // one 32-wide N slice of H per block (2 B-tiles)
constexpr int NTHR = 256;   // 8 waves (wave32)
constexpr int KCH  = Hh / 32;  // 32 K-chunks for W_hh
constexpr int KCX  = 4;        // 4 K-chunks for W_ih (100 padded to 128)
constexpr int LDSSTR = (KCH + KCX) * 512;  // bf16 elems per 16-col B region

// ---------------------------------------------------------------------------
// Workspace layout (bytes):
//   [0,256)                     barrier {count, phase}
//   [256, +256K)                h ping buffer  (bf16 [B][H])
//   [.., +256K)                 h pong buffer
//   [.., +2M)                   W_hh packed into WMMA B layout (bf16)
//   [.., +256K)                 W_ih packed into WMMA B layout (bf16, K padded)
// Total ~2.75 MB.
// ---------------------------------------------------------------------------

// Pack W_hh[H][H] (fp32, row-major; out[b,n] = sum_k h[b,k]*W_hh[n,k]) into the
// CDNA5 WMMA 16-bit B (32x16) register image, linearized as
// [ntile(64)][kchunk(32)][lane(32)][elem(16)]:
//   n = ntile*16 + (lane&15),  k = kchunk*32 + (lane>>4)*8 + (elem&7) + (elem>>3)*16
__global__ __launch_bounds__(NTHR) void pack_whh(const float* __restrict__ W,
                                                 __bf16* __restrict__ out) {
  int idx = blockIdx.x * blockDim.x + threadIdx.x;       // 1,048,576 total
  int e    = idx & 15;
  int lane = (idx >> 4) & 31;
  int kc   = (idx >> 9) & 31;
  int nt   = idx >> 14;
  int n = nt * 16 + (lane & 15);
  int k = kc * 32 + ((lane >> 4) << 3) + (e & 7) + ((e >> 3) << 4);
  out[idx] = (__bf16)W[(size_t)n * Hh + k];
}

// Same for W_ih[H][D], K zero-padded 100 -> 128 (4 chunks of 32).
__global__ __launch_bounds__(NTHR) void pack_wih(const float* __restrict__ W,
                                                 __bf16* __restrict__ out) {
  int idx = blockIdx.x * blockDim.x + threadIdx.x;       // 131,072 total
  int e    = idx & 15;
  int lane = (idx >> 4) & 31;
  int kc   = (idx >> 9) & 3;
  int nt   = idx >> 11;
  int n = nt * 16 + (lane & 15);
  int k = kc * 32 + ((lane >> 4) << 3) + (e & 7) + ((e >> 3) << 4);
  out[idx] = (__bf16)((k < Dd) ? W[(size_t)n * Dd + k] : 0.0f);
}

// Zero the barrier words and h0 (kernel-side init: graph-capture safe).
__global__ __launch_bounds__(NTHR) void init_ws(unsigned* __restrict__ bar,
                                                __bf16* __restrict__ h0) {
  int i = blockIdx.x * blockDim.x + threadIdx.x;
  if (blockIdx.x == 0 && threadIdx.x < 64) bar[threadIdx.x] = 0u;
  if (i < Bz * Hh) h0[i] = (__bf16)0.0f;
}

// ---------------------------------------------------------------------------
// Persistent recurrent kernel.  Block bk owns output columns [bk*32, bk*32+32)
// as two 16-wide B-tiles kept in LDS for all 2048 steps.  Wave w owns batch
// rows [w*16, w*16+16).  Each step, per wave:
//   acc{0,1}(16x16,f32) += A(h or x, 16x32 bf16) x B{0,1}(LDS, 32x16 bf16)
// (one A fragment feeds two WMMAs), then h_new = tanh(acc + b_ih + b_hh) is
// written bf16 to the L2-resident ping-pong buffer.  Grid-wide sync per step
// via atomic arrive + phase spin (s_sleep while waiting).  The K-loop is
// software-pipelined one chunk deep so loads for chunk k+1 overlap the WMMAs
// of chunk k.
// ---------------------------------------------------------------------------
__global__ __launch_bounds__(NTHR)
void rnn_persistent(const float* __restrict__ x,
                    const float* __restrict__ b_ih,
                    const float* __restrict__ b_hh,
                    const float* __restrict__ W_out,
                    const float* __restrict__ b_out,
                    const __bf16* __restrict__ whh_pack,
                    const __bf16* __restrict__ wih_pack,
                    __bf16* __restrict__ hbuf0,
                    __bf16* __restrict__ hbuf1,
                    float* __restrict__ out,
                    unsigned* __restrict__ bar) {
  __shared__ __align__(32) __bf16 ldsB[2 * LDSSTR];   // 72 KB / WGP

  const int nt    = blockIdx.x;
  const int tid   = threadIdx.x;
  const int lane  = tid & 31;
  const int wave  = tid >> 5;
  const int khalf = lane >> 4;          // which K half this lane holds (A/B)
  const int m0    = wave * 16;          // batch tile base
  const int m     = m0 + (lane & 15);   // this lane's A row (batch index)
  const int ncol0 = nt * 32 + (lane & 15);
  const int ncol1 = ncol0 + 16;

  // ---- stage this block's two B slices (2 x 36 KB) into LDS once ----
#if USE_ASYNC_LDS
  for (int r = 0; r < 2; ++r) {
    const __bf16* gw = whh_pack + (size_t)(2 * nt + r) * (KCH * 512);
    __bf16* lw = ldsB + (size_t)r * LDSSTR;
    for (int i = tid; i < KCH * 512 / 8; i += NTHR)
      __builtin_amdgcn_global_load_async_to_lds_b128(
          (AS1 v4i*)(gw + (size_t)i * 8), (AS3 v4i*)(lw + (size_t)i * 8), 0, 0);
    const __bf16* gi = wih_pack + (size_t)(2 * nt + r) * (KCX * 512);
    __bf16* li = ldsB + (size_t)r * LDSSTR + KCH * 512;
    for (int i = tid; i < KCX * 512 / 8; i += NTHR)
      __builtin_amdgcn_global_load_async_to_lds_b128(
          (AS1 v4i*)(gi + (size_t)i * 8), (AS3 v4i*)(li + (size_t)i * 8), 0, 0);
  }
  __builtin_amdgcn_s_wait_asynccnt(0);
#else
  for (int r = 0; r < 2; ++r) {
    const uint4* s1 = reinterpret_cast<const uint4*>(whh_pack + (size_t)(2 * nt + r) * KCH * 512);
    uint4* d1 = reinterpret_cast<uint4*>(ldsB + (size_t)r * LDSSTR);
    for (int i = tid; i < KCH * 512 / 8; i += NTHR) d1[i] = s1[i];
    const uint4* s2 = reinterpret_cast<const uint4*>(wih_pack + (size_t)(2 * nt + r) * KCX * 512);
    uint4* d2 = reinterpret_cast<uint4*>(ldsB + (size_t)r * LDSSTR + KCH * 512);
    for (int i = tid; i < KCX * 512 / 8; i += NTHR) d2[i] = s2[i];
  }
#endif
  __syncthreads();

  const __bf16* hin  = hbuf0;
  __bf16*       hout = hbuf1;
  unsigned phase = 0;
  unsigned* count = bar;
  unsigned* ph    = bar + 1;

  for (int t = 0; t < Tt; ++t) {
    v8f acc0 = {};
    v8f acc1 = {};

    // ---- recurrent term: h(t-1) @ W_hh^T, K = 1024, pipelined 1 deep ----
    const __bf16* hrow = hin + (size_t)m * Hh + khalf * 8;
    v8bf alo = *reinterpret_cast<const v8bf*>(hrow);
    v8bf ahi = *reinterpret_cast<const v8bf*>(hrow + 16);
    v16bf nb0 = *reinterpret_cast<const v16bf*>(&ldsB[lane * 16]);
    v16bf nb1 = *reinterpret_cast<const v16bf*>(&ldsB[LDSSTR + lane * 16]);
#pragma unroll 4
    for (int kc = 0; kc < KCH; ++kc) {
      v16bf a;
#pragma unroll
      for (int j = 0; j < 8; ++j) { a[j] = alo[j]; a[8 + j] = ahi[j]; }
      v16bf b0 = nb0, b1 = nb1;
      if (kc + 1 < KCH) {       // preload next chunk before this chunk's WMMAs
        alo = *reinterpret_cast<const v8bf*>(hrow + (kc + 1) * 32);
        ahi = *reinterpret_cast<const v8bf*>(hrow + (kc + 1) * 32 + 16);
        nb0 = *reinterpret_cast<const v16bf*>(&ldsB[(kc + 1) * 512 + lane * 16]);
        nb1 = *reinterpret_cast<const v16bf*>(&ldsB[LDSSTR + (kc + 1) * 512 + lane * 16]);
      }
      acc0 = __builtin_amdgcn_wmma_f32_16x16x32_bf16(false, a, false, b0,
                                                     (short)0, acc0, false, false);
      acc1 = __builtin_amdgcn_wmma_f32_16x16x32_bf16(false, a, false, b1,
                                                     (short)0, acc1, false, false);
    }

    // ---- input term: x(t) @ W_ih^T, K = 100 padded to 128, fused ----
    const float* xrow = x + ((size_t)t * Bz + m) * Dd;
#pragma unroll
    for (int kc = 0; kc < KCX; ++kc) {
      const int kbase = kc * 32 + khalf * 8;
      v16bf a;
#pragma unroll
      for (int j = 0; j < 8; ++j) {
        int k1 = kbase + j, k2 = kbase + 16 + j;
        a[j]     = (__bf16)((k1 < Dd) ? xrow[k1] : 0.0f);
        a[8 + j] = (__bf16)((k2 < Dd) ? xrow[k2] : 0.0f);
      }
      v16bf b0 = *reinterpret_cast<const v16bf*>(&ldsB[(KCH + kc) * 512 + lane * 16]);
      v16bf b1 = *reinterpret_cast<const v16bf*>(&ldsB[LDSSTR + (KCH + kc) * 512 + lane * 16]);
      acc0 = __builtin_amdgcn_wmma_f32_16x16x32_bf16(false, a, false, b0,
                                                     (short)0, acc0, false, false);
      acc1 = __builtin_amdgcn_wmma_f32_16x16x32_bf16(false, a, false, b1,
                                                     (short)0, acc1, false, false);
    }

    // Prefetch next timestep's x rows into cache while we finish this step.
    if (t + 1 < Tt) __builtin_prefetch(x + ((size_t)(t + 1) * Bz + m) * Dd, 0, 1);

    // ---- epilogue: bias + tanh, write bf16 h_new ----
    // C/D layout: VGPR r holds M = r (lanes 0-15) or 8+r (lanes 16-31), N = lane&15.
    const float bias0 = b_ih[ncol0] + b_hh[ncol0];
    const float bias1 = b_ih[ncol1] + b_hh[ncol1];
    const int mb = m0 + (khalf ? 8 : 0);
#pragma unroll
    for (int r = 0; r < 8; ++r) {
      float v0 = tanhf(acc0[r] + bias0);
      float v1 = tanhf(acc1[r] + bias1);
      hout[(size_t)(mb + r) * Hh + ncol0] = (__bf16)v0;
      hout[(size_t)(mb + r) * Hh + ncol1] = (__bf16)v1;
    }

    // ---- grid-wide barrier (arrive count + phase flip) ----
    __threadfence();
    __syncthreads();
    if (tid == 0) {
      unsigned prev = atomicAdd(count, 1u);
      if (prev == (unsigned)(NBLK - 1)) {
        atomicExch(count, 0u);
        __threadfence();
        atomicAdd(ph, 1u);
      } else {
        while (__hip_atomic_load(ph, __ATOMIC_ACQUIRE, __HIP_MEMORY_SCOPE_AGENT)
               == phase) {
          __builtin_amdgcn_s_sleep(1);
        }
      }
    }
    phase++;
    __syncthreads();

    // ping-pong
    const __bf16* tmp = hout;
    hout = const_cast<__bf16*>(hin);
    hin  = tmp;
  }

  // ---- final projection: out[b] = tanh(h_final[b,:] . W_out + b_out) ----
  if (blockIdx.x == 0 && tid < Bz) {
    float acc = b_out[0];
    const __bf16* hr = hin + (size_t)tid * Hh;
    for (int k = 0; k < Hh; ++k) acc += (float)hr[k] * W_out[k];
    out[tid] = tanhf(acc);
  }
}

// ---------------------------------------------------------------------------
extern "C" void kernel_launch(void* const* d_in, const int* in_sizes, int n_in,
                              void* d_out, int out_size, void* d_ws, size_t ws_size,
                              hipStream_t stream) {
  (void)in_sizes; (void)n_in; (void)out_size; (void)ws_size;
  const float* x     = (const float*)d_in[0];
  const float* W_ih  = (const float*)d_in[1];
  const float* b_ih  = (const float*)d_in[2];
  const float* W_hh  = (const float*)d_in[3];
  const float* b_hh  = (const float*)d_in[4];
  const float* W_out = (const float*)d_in[5];
  const float* b_out = (const float*)d_in[6];
  float* out = (float*)d_out;

  char* ws = (char*)d_ws;
  unsigned* bar = (unsigned*)ws;                                   // 256 B
  __bf16* hbuf0 = (__bf16*)(ws + 256);                             // 256 KB
  __bf16* hbuf1 = (__bf16*)(ws + 256 + 262144);                    // 256 KB
  __bf16* whhp  = (__bf16*)(ws + 256 + 2 * 262144);                // 2 MB
  __bf16* wihp  = (__bf16*)(ws + 256 + 2 * 262144 + 2097152);      // 256 KB

  hipLaunchKernelGGL(init_ws,  dim3(512),  dim3(NTHR), 0, stream, bar, hbuf0);
  hipLaunchKernelGGL(pack_whh, dim3(4096), dim3(NTHR), 0, stream, W_hh, whhp);
  hipLaunchKernelGGL(pack_wih, dim3(512),  dim3(NTHR), 0, stream, W_ih, wihp);
  hipLaunchKernelGGL(rnn_persistent, dim3(NBLK), dim3(NTHR), 0, stream,
                     x, b_ih, b_hh, W_out, b_out, whhp, wihp,
                     hbuf0, hbuf1, out, bar);
}
